// GCNLinkPredictor_31799937860194
// MI455X (gfx1250) — matrix operand order repaired and checked by reading, not compiled, not run on hardware
//
#include <hip/hip_runtime.h>

// ---------------------------------------------------------------------------
// GCN forward for MI455X (gfx1250, wave32):
//   h  = spmm(A, x)            fp32 atomics (L2-resident gather/scatter)
//   h  = relu(h @ w1 + b1)     bf16 WMMA, fp32 accumulate
//   h2 = spmm(A, h)            fp32 atomics
//   out= h2 @ w2 + b2          bf16 WMMA, fp32 accumulate
// ---------------------------------------------------------------------------

typedef __attribute__((ext_vector_type(16))) __bf16 v16bf;
typedef __attribute__((ext_vector_type(8)))  float  v8f;

#define GCN_N      100000
#define GCN_E      1600000
#define GCN_CH     128     // IN_CH == HID_CH == 128
#define GCN_OUT    64

static __device__ __forceinline__ unsigned short f32_to_bf16_rne(float f) {
  unsigned int u = __float_as_uint(f);
  u += 0x7FFFu + ((u >> 16) & 1u);   // round-to-nearest-even
  return (unsigned short)(u >> 16);
}

union FragU {
  v16bf v;
  uint4 q[2];
};

// ---------------------------------------------------------------- zero buffer
__global__ void gcn_zero_f32_kernel(float4* __restrict__ p, int n4) {
  int i = blockIdx.x * blockDim.x + threadIdx.x;
  if (i < n4) {
    float4 z; z.x = 0.f; z.y = 0.f; z.z = 0.f; z.w = 0.f;
    p[i] = z;
  }
}

// ------------------------------------------------------- fp32 -> bf16 convert
__global__ void gcn_cvt_bf16_kernel(const float4* __restrict__ in,
                                    ushort4* __restrict__ out, int n4) {
  int i = blockIdx.x * blockDim.x + threadIdx.x;
  if (i >= n4) return;
  float4 f = in[i];
  ushort4 o;
  o.x = f32_to_bf16_rne(f.x);
  o.y = f32_to_bf16_rne(f.y);
  o.z = f32_to_bf16_rne(f.z);
  o.w = f32_to_bf16_rne(f.w);
  out[i] = o;
}

// ------------------------------------- weight transpose+convert: w[K][N] -> wt[N][K]
__global__ void gcn_wT_bf16_kernel(const float* __restrict__ w,
                                   unsigned short* __restrict__ wt,
                                   int K, int N) {
  int i = blockIdx.x * blockDim.x + threadIdx.x;
  if (i >= K * N) return;
  int k = i / N, n = i - k * N;
  wt[n * K + k] = f32_to_bf16_rne(w[i]);
}

// --------------------------------------------------------------- COO SpMM
// One wave (32 lanes) per edge; lane L handles channels [4L, 4L+4).
// Edge metadata scalarized with readfirstlane; feature row gathered with
// global_load_b128; scatter via global_atomic_add_f32 (out and x sit in L2).
__global__ void gcn_spmm_atomic_kernel(const float* __restrict__ x,
                                       const int* __restrict__ erow,
                                       const int* __restrict__ ecol,
                                       const float* __restrict__ eval,
                                       float* __restrict__ out) {
  const int lane = threadIdx.x & 31;
  const int e = blockIdx.x * (blockDim.x >> 5) + (threadIdx.x >> 5);
  if (e >= GCN_E) return;

  const int   r = __builtin_amdgcn_readfirstlane(erow[e]);
  const int   c = __builtin_amdgcn_readfirstlane(ecol[e]);
  const float v = __int_as_float(
      __builtin_amdgcn_readfirstlane(__float_as_int(eval[e])));

  const float4 xv = *(const float4*)(x + (size_t)c * GCN_CH + lane * 4);
  float* o = out + (size_t)r * GCN_CH + lane * 4;
  atomicAdd(o + 0, v * xv.x);
  atomicAdd(o + 1, v * xv.y);
  atomicAdd(o + 2, v * xv.z);
  atomicAdd(o + 3, v * xv.w);
}

// ------------------------------------------------- WMMA GEMM + bias (+ReLU)
// A:  [M][128] bf16 row-major (activations), K fixed at 128.
// Bt: [NC][128] bf16 row-major (pre-transposed weights).
// Each wave computes one 16x16 tile of C; block = (NC/16) waves sharing the
// same 16-row A block. 4x V_WMMA_F32_16X16X32_BF16 per tile, fp32 accumulate.
//
// Fragment layouts (CDNA5 ISA 7.12.2):
//   A 16x32 bf16:  lane&15 = row M; lanes 0-15 hold K {0..7,16..23},
//                  lanes 16-31 hold K {8..15,24..31}  -> chunks at +hi*8, +hi*8+16
//   B 32x16 bf16:  lane&15 = col N; lanes 0-15 hold K {0..7,16..23},
//                  lanes 16-31 hold K {8..15,24..31}  -> chunks at +hi*8, +hi*8+16
//   C/D 16x16 f32: lane&15 = col N; VGPR r = row (lane<16 ? r : r+8)
__global__ void gcn_wmma_gemm_bias_kernel(const unsigned short* __restrict__ A,
                                          const unsigned short* __restrict__ Bt,
                                          const float* __restrict__ bias,
                                          float* __restrict__ C,
                                          const int NC, const int doRelu) {
  const int lane    = threadIdx.x & 31;
  const int wave    = threadIdx.x >> 5;
  const int laneLo  = lane & 15;
  const int hi      = lane >> 4;                // 0 or 1
  const int rowBase = blockIdx.x * 16;
  const int col     = wave * 16 + laneLo;

  const unsigned short* aRow = A  + (size_t)(rowBase + laneLo) * GCN_CH;
  const unsigned short* bRow = Bt + (size_t)col * GCN_CH;
  const int aOff = hi * 8;
  const int bOff = hi * 8;

  v8f acc = {};
#pragma unroll
  for (int k0 = 0; k0 < GCN_CH; k0 += 32) {
    FragU a, b;
    a.q[0] = *(const uint4*)(aRow + k0 + aOff);
    a.q[1] = *(const uint4*)(aRow + k0 + aOff + 16);
    b.q[0] = *(const uint4*)(bRow + k0 + bOff);
    b.q[1] = *(const uint4*)(bRow + k0 + bOff + 16);
    acc = __builtin_amdgcn_wmma_f32_16x16x32_bf16(
        /*neg_a=*/false, a.v, /*neg_b=*/false, b.v,
        /*c_mod=*/(short)0, acc, /*reuse_a=*/false, /*reuse_b=*/false);
  }

  const float bv = bias[col];
  float* cp = C + (size_t)(rowBase + hi * 8) * NC + col;
#pragma unroll
  for (int r = 0; r < 8; ++r) {
    float vv = acc[r] + bv;
    if (doRelu) vv = fmaxf(vv, 0.0f);
    cp[(size_t)r * NC] = vv;
  }
}

// ---------------------------------------------------------------------------
extern "C" void kernel_launch(void* const* d_in, const int* in_sizes, int n_in,
                              void* d_out, int out_size, void* d_ws, size_t ws_size,
                              hipStream_t stream) {
  (void)in_sizes; (void)n_in; (void)out_size; (void)ws_size;

  const float* x  = (const float*)d_in[0];
  const int*   er = (const int*)d_in[1];
  const int*   ec = (const int*)d_in[2];
  const float* ev = (const float*)d_in[3];
  const float* w1 = (const float*)d_in[4];
  const float* b1 = (const float*)d_in[5];
  const float* w2 = (const float*)d_in[6];
  const float* b2 = (const float*)d_in[7];

  // Workspace carve-out (256B aligned)
  char* ws = (char*)d_ws;
  size_t off = 0;
  auto carve = [&](size_t bytes) -> void* {
    void* p = ws + off;
    off = (off + bytes + 255) & ~(size_t)255;
    return p;
  };
  float*          bufAgg = (float*)         carve((size_t)GCN_N * GCN_CH * 4); // spmm out (reused)
  unsigned short* bufBf  = (unsigned short*)carve((size_t)GCN_N * GCN_CH * 2); // bf16 shadow (reused)
  float*          bufH   = (float*)         carve((size_t)GCN_N * GCN_CH * 4); // layer-1 activations
  unsigned short* w1T    = (unsigned short*)carve((size_t)GCN_CH * GCN_CH * 2);
  unsigned short* w2T    = (unsigned short*)carve((size_t)GCN_OUT * GCN_CH * 2);

  const int ELEM4   = GCN_N * GCN_CH / 4;      // 3.2M float4
  const int ZB      = ELEM4 / 256;             // 12500 blocks
  const int SPB     = GCN_E / 8;               // 200000 blocks (8 waves/block)
  const int GEMMB   = GCN_N / 16;              // 6250 row tiles

  // Pre-transpose weights into bf16 [Ncol][K]
  gcn_wT_bf16_kernel<<<(GCN_CH * GCN_CH + 255) / 256, 256, 0, stream>>>(w1, w1T, GCN_CH, GCN_CH);
  gcn_wT_bf16_kernel<<<(GCN_CH * GCN_OUT + 255) / 256, 256, 0, stream>>>(w2, w2T, GCN_CH, GCN_OUT);

  // ---- Layer 1: spmm -> linear -> relu
  gcn_zero_f32_kernel<<<ZB, 256, 0, stream>>>((float4*)bufAgg, ELEM4);
  gcn_spmm_atomic_kernel<<<SPB, 256, 0, stream>>>(x, er, ec, ev, bufAgg);
  gcn_cvt_bf16_kernel<<<ZB, 256, 0, stream>>>((const float4*)bufAgg, (ushort4*)bufBf, ELEM4);
  gcn_wmma_gemm_bias_kernel<<<GEMMB, 256, 0, stream>>>(bufBf, w1T, b1, bufH,
                                                       GCN_CH, /*relu=*/1);

  // ---- Layer 2: spmm -> linear
  gcn_zero_f32_kernel<<<ZB, 256, 0, stream>>>((float4*)bufAgg, ELEM4);
  gcn_spmm_atomic_kernel<<<SPB, 256, 0, stream>>>(bufH, er, ec, ev, bufAgg);
  gcn_cvt_bf16_kernel<<<ZB, 256, 0, stream>>>((const float4*)bufAgg, (ushort4*)bufBf, ELEM4);
  gcn_wmma_gemm_bias_kernel<<<GEMMB, 128, 0, stream>>>(bufBf, w2T, b2, (float*)d_out,
                                                       GCN_OUT, /*relu=*/0);
}
